// IGNN_26594437497118
// MI455X (gfx1250) — compile-verified, matrix operand order checked by reading, not run on hardware
//
#include <hip/hip_runtime.h>
#include <hip/hip_bf16.h>
#include <cstdint>

typedef _Float16 v16h __attribute__((ext_vector_type(16)));
typedef _Float16 v8h  __attribute__((ext_vector_type(8)));
typedef _Float16 v4h  __attribute__((ext_vector_type(4)));
typedef float    v8f  __attribute__((ext_vector_type(8)));

#define NN 25000
#define NE 400000
#define F   32      // unified feature width (node/edge feats padded to 32)
#define DH  64      // MLP hidden width
#define EDIN 96     // edge MLP input (xs|xd|e), padded
#define NDIN 64     // node MLP input (x|agg), padded
#define ROWS 128    // rows per block (8 waves * 16)

// f16 weight arena (element offsets); weights stored pre-swizzled as A-fragments of W^T
#define LAYER_W_STRIDE 30720
#define EW0_OFF 0
#define EW1_OFF 6144
#define EW2_OFF 10240
#define EW3_OFF 14336
#define NW_BASE 16384   // node W0 +0, W1 +4096, W2 +8192, W3 +12288
#define LAYER_B_STRIDE 448   // floats: edge biases at +0 (224), node at +224

struct PP { const float* p[64]; };   // per layer: eW0..3, eB0..3, nW0..3, nB0..3

// ---------------- weight conversion: pad + transpose + A-fragment swizzle ----------------
// A = W^T fragment (mt, kc): lane L, half h holds W[k][col] with
//   col = mt*16 + (L&15),  k = kc*32 + ((h&8)?16:0) + (L>>4)*8 + (h&7)
// storage offset = ((mt*KC + kc)*32 + L)*16 + h   (contiguous 32B per lane)
__global__ void conv_weights_kernel(PP pp, _Float16* __restrict__ wsW,
                                    float* __restrict__ wsB) {
  int b = blockIdx.x;
  if (b < 32) {
    int l = b >> 3, m = b & 7;
    const int padK[8]   = {EDIN, DH, DH, DH, NDIN, DH, DH, DH};
    const int padN[8]   = {DH, DH, DH, F, DH, DH, DH, F};
    const int dstOff[8] = {EW0_OFF, EW1_OFF, EW2_OFF, EW3_OFF,
                           NW_BASE + 0, NW_BASE + 4096, NW_BASE + 8192, NW_BASE + 12288};
    int K = padK[m], Nd = padN[m];
    int KC = K >> 5;
    int pidx = l * 16 + ((m < 4) ? m : (8 + (m - 4)));
    const float* srcp = pp.p[pidx];
    _Float16* dstp = wsW + (size_t)l * LAYER_W_STRIDE + dstOff[m];
    int total = K * Nd;
    for (int o = threadIdx.x; o < total; o += blockDim.x) {
      int fi   = o >> 9;           // fragment index (512 halves each)
      int rem  = o & 511;
      int lane = rem >> 4;
      int h    = rem & 15;
      int mt = fi / KC, kc = fi % KC;
      int nl = lane & 15, g = lane >> 4;
      int col = mt * 16 + nl;                                   // W column (out feature)
      int k   = kc * 32 + ((h & 8) ? 16 : 0) + g * 8 + (h & 7); // padded W row (in feature)
      int sr;
      if (m == 0 && l == 0) {          // edge l0: src rows 0-15 xs, 16-31 xd, 32-39 e
        sr = (k < 16) ? k : (k >= 32 && k < 48) ? (k - 16)
           : (k >= 64 && k < 72) ? (k - 32) : -1;
      } else if (m == 4 && l == 0) {   // node l0: src rows 0-15 x, 16-47 agg
        sr = (k < 16) ? k : (k >= 32 && k < 64) ? (k - 16) : -1;
      } else {
        sr = (k < K) ? k : -1;
      }
      float v = (sr >= 0) ? srcp[(size_t)sr * Nd + col] : 0.0f;
      dstp[o] = (_Float16)v;
    }
  } else {
    int b2 = b - 32; int l = b2 >> 1, side = b2 & 1;
    const int bsz[4]  = {DH, DH, DH, F};
    const int boff[4] = {0, 64, 128, 192};
    for (int j = 0; j < 4; ++j) {
      const float* src = pp.p[l * 16 + (side ? 12 : 4) + j];
      float* dst = wsB + (size_t)l * LAYER_B_STRIDE + side * 224 + boff[j];
      for (int i = threadIdx.x; i < bsz[j]; i += blockDim.x) dst[i] = src[i];
    }
  }
}

__global__ void pad_feat_kernel(const float* __restrict__ src, float* __restrict__ dst,
                                long rows, int srcCols) {
  long i = (long)blockIdx.x * blockDim.x + threadIdx.x;
  long total = rows * F;
  if (i >= total) return;
  long r = i / F; int c = (int)(i % F);
  dst[i] = (c < srcCols) ? src[r * srcCols + c] : 0.0f;
}

__global__ void zero_kernel(float* __restrict__ p, long n) {
  long i = (long)blockIdx.x * blockDim.x + threadIdx.x;
  if (i < n) p[i] = 0.0f;
}

// ---------------- fragment helpers ----------------
__device__ __forceinline__ v16h load_frag(const _Float16* p0, const _Float16* p1) {
  v8h lo = *(const v8h*)p0;   // 16B vector load (b128)
  v8h hi = *(const v8h*)p1;
  return __builtin_shufflevector(lo, hi, 0, 1, 2, 3, 4, 5, 6, 7,
                                 8, 9, 10, 11, 12, 13, 14, 15);
}

// out^T GEMM step: A = pre-swizzled W^T fragments (global, L2-hot), B = activation frags
template <int MT, int KC>
__device__ __forceinline__ void gemmT(const _Float16* __restrict__ wmat, int lane,
                                      const v16h* bf, v8f* acc) {
#pragma unroll
  for (int mt = 0; mt < MT; ++mt) {
    v8f c;
#pragma unroll
    for (int j = 0; j < 8; ++j) c[j] = 0.0f;
#pragma unroll
    for (int kc = 0; kc < KC; ++kc) {
      const _Float16* p = wmat + (size_t)(((mt * KC + kc) << 5) + lane) * 16;
      v16h a = load_frag(p, p + 8);
      c = __builtin_amdgcn_wmma_f32_16x16x32_f16(false, a, false, bf[kc], (short)0,
                                                 c, false, false);
    }
    acc[mt] = c;
  }
}

// Repack C accumulators (out^T layout) into B fragments of the next GEMM:
// B frag kc, half j   <- acc[2kc][j]   (feature kc*32 + 8g + j)
//           half 8+j  <- acc[2kc+1][j] (feature kc*32 + 16 + 8g + j)
// with bias + ReLU fused. Pure registers, no LDS, no cross-lane movement.
template <int KCN>
__device__ __forceinline__ void build_b(const v8f* acc, const float* __restrict__ bias,
                                        int g, v16h* out) {
#pragma unroll
  for (int kc = 0; kc < KCN; ++kc) {
    v16h o;
#pragma unroll
    for (int j = 0; j < 8; ++j) {
      float v = acc[2 * kc][j]     + bias[kc * 32 + 8 * g + j];
      float w = acc[2 * kc + 1][j] + bias[kc * 32 + 16 + 8 * g + j];
      o[j]     = (_Float16)fmaxf(v, 0.0f);
      o[8 + j] = (_Float16)fmaxf(w, 0.0f);
    }
    out[kc] = o;
  }
}

// ---------------- edge update ----------------
__global__ void __launch_bounds__(256)
edge_mlp_kernel(const float* __restrict__ x, const float* __restrict__ e,
                const long long* __restrict__ src, const long long* __restrict__ dst,
                const _Float16* __restrict__ gW, const float* __restrict__ gB,
                float* __restrict__ eout, float* __restrict__ agg) {
  __shared__ alignas(16) _Float16 smA[ROWS * EDIN];  // 24 KB
  __shared__ int smSrc[ROWS], smDst[ROWS];

  const int tid = threadIdx.x;
  const int lane = tid & 31, wave = tid >> 5;
  const long e0 = (long)blockIdx.x * ROWS;
  const int waveBase = wave * 16;

  if (tid < 4) {  // hint weight arena into cache (global_prefetch_b8)
    const int woff[4] = {EW0_OFF, EW1_OFF, EW2_OFF, EW3_OFF};
    __builtin_prefetch(gW + woff[tid], 0, 1);
  }
  for (int r = tid; r < ROWS; r += 256) {
    long row = e0 + r; if (row >= NE) row = NE - 1;
    smSrc[r] = (int)src[row];
    smDst[r] = (int)dst[row];
  }
  __syncthreads();
  // stage A^T source rows: [x_src | x_dst | e] as f16, float4 -> v4h packed stores
  for (int i = tid; i < ROWS * (EDIN / 4); i += 256) {
    int r = i / (EDIN / 4), cg = i % (EDIN / 4);
    int c = cg * 4;
    long row = e0 + r; if (row >= NE) row = NE - 1;
    float4 f;
    if (c < F)            f = *(const float4*)(x + (long)smSrc[r] * F + c);
    else if (c < 2 * F)   f = *(const float4*)(x + (long)smDst[r] * F + (c - F));
    else                  f = *(const float4*)(e + row * F + (c - 2 * F));
    v4h h4;
    h4[0] = (_Float16)f.x; h4[1] = (_Float16)f.y;
    h4[2] = (_Float16)f.z; h4[3] = (_Float16)f.w;
    *(v4h*)(smA + r * EDIN + c) = h4;
  }
  __syncthreads();

  const int g = lane >> 4, nl = lane & 15;
  const int myrow = waveBase + nl;

  // first-layer B fragments from staged inputs (2x ds_load_b128 each)
  v16h bf[3];
#pragma unroll
  for (int kc = 0; kc < 3; ++kc) {
    const _Float16* p = smA + myrow * EDIN + kc * 32 + g * 8;
    bf[kc] = load_frag(p, p + 16);
  }

  v8f acc[4];
  gemmT<4, 3>(gW + EW0_OFF, lane, bf, acc);
  v16h bh[2];
  build_b<2>(acc, gB + 0, g, bh);
  gemmT<4, 2>(gW + EW1_OFF, lane, bh, acc);
  build_b<2>(acc, gB + 64, g, bh);
  gemmT<4, 2>(gW + EW2_OFF, lane, bh, acc);
  build_b<2>(acc, gB + 128, g, bh);
  v8f ao[2];
  gemmT<2, 2>(gW + EW3_OFF, lane, bh, ao);

  // epilogue: lane owns edge row (e0+myrow), features [8g,8g+8) and [16+8g,16+8g+8)
  const float* b3 = gB + 192;
  float v0[8], v1[8], ss = 0.0f;
#pragma unroll
  for (int j = 0; j < 8; ++j) {
    v0[j] = ao[0][j] + b3[8 * g + j];
    v1[j] = ao[1][j] + b3[16 + 8 * g + j];
    ss += v0[j] * v0[j] + v1[j] * v1[j];
  }
  ss += __shfl_xor(ss, 16);           // partner lane holds the other 16 features
  float inv = rsqrtf(ss);
  long row = e0 + myrow;
  if (row < NE) {
    float4 s0 = {v0[0] * inv, v0[1] * inv, v0[2] * inv, v0[3] * inv};
    float4 s1 = {v0[4] * inv, v0[5] * inv, v0[6] * inv, v0[7] * inv};
    float4 s2 = {v1[0] * inv, v1[1] * inv, v1[2] * inv, v1[3] * inv};
    float4 s3 = {v1[4] * inv, v1[5] * inv, v1[6] * inv, v1[7] * inv};
    *(float4*)(eout + row * F + 8 * g)          = s0;
    *(float4*)(eout + row * F + 8 * g + 4)      = s1;
    *(float4*)(eout + row * F + 16 + 8 * g)     = s2;
    *(float4*)(eout + row * F + 16 + 8 * g + 4) = s3;
    long d = (long)smDst[myrow];
#pragma unroll
    for (int j = 0; j < 8; ++j) {
      atomicAdd(&agg[d * F + 8 * g + j], v0[j]);
      atomicAdd(&agg[d * F + 16 + 8 * g + j], v1[j]);
    }
  }
}

// ---------------- node update ----------------
__global__ void __launch_bounds__(256)
node_mlp_kernel(const float* __restrict__ x, const float* __restrict__ agg,
                const float* __restrict__ deg,
                const _Float16* __restrict__ gW, const float* __restrict__ gB,
                float* __restrict__ xout) {
  __shared__ alignas(16) _Float16 smA[ROWS * NDIN];  // 16 KB
  __shared__ float smInv[ROWS];

  const int tid = threadIdx.x;
  const int lane = tid & 31, wave = tid >> 5;
  const long n0 = (long)blockIdx.x * ROWS;
  const int waveBase = wave * 16;

  if (tid < 4) {
    const int woff[4] = {0, 4096, 8192, 12288};
    __builtin_prefetch(gW + woff[tid], 0, 1);
  }
  for (int r = tid; r < ROWS; r += 256) {
    long row = n0 + r; if (row >= NN) row = NN - 1;
    smInv[r] = 1.0f / deg[row];
  }
  __syncthreads();
  for (int i = tid; i < ROWS * (NDIN / 4); i += 256) {
    int r = i / (NDIN / 4), cg = i % (NDIN / 4);
    int c = cg * 4;
    long row = n0 + r; if (row >= NN) row = NN - 1;
    float4 f;
    if (c < F) {
      f = *(const float4*)(x + row * F + c);
    } else {
      f = *(const float4*)(agg + row * F + (c - F));
      float iv = smInv[r];
      f.x *= iv; f.y *= iv; f.z *= iv; f.w *= iv;
    }
    v4h h4;
    h4[0] = (_Float16)f.x; h4[1] = (_Float16)f.y;
    h4[2] = (_Float16)f.z; h4[3] = (_Float16)f.w;
    *(v4h*)(smA + r * NDIN + c) = h4;
  }
  __syncthreads();

  const int g = lane >> 4, nl = lane & 15;
  const int myrow = waveBase + nl;

  v16h bf[2];
#pragma unroll
  for (int kc = 0; kc < 2; ++kc) {
    const _Float16* p = smA + myrow * NDIN + kc * 32 + g * 8;
    bf[kc] = load_frag(p, p + 16);
  }

  v8f acc[4];
  gemmT<4, 2>(gW + 0, lane, bf, acc);
  v16h bh[2];
  build_b<2>(acc, gB + 0, g, bh);
  gemmT<4, 2>(gW + 4096, lane, bh, acc);
  build_b<2>(acc, gB + 64, g, bh);
  gemmT<4, 2>(gW + 8192, lane, bh, acc);
  build_b<2>(acc, gB + 128, g, bh);
  v8f ao[2];
  gemmT<2, 2>(gW + 12288, lane, bh, ao);

  const float* b3 = gB + 192;
  float v0[8], v1[8], ss = 0.0f;
#pragma unroll
  for (int j = 0; j < 8; ++j) {
    v0[j] = ao[0][j] + b3[8 * g + j];
    v1[j] = ao[1][j] + b3[16 + 8 * g + j];
    ss += v0[j] * v0[j] + v1[j] * v1[j];
  }
  ss += __shfl_xor(ss, 16);
  float inv = rsqrtf(ss);
  long row = n0 + myrow;
  if (row < NN) {
    float4 s0 = {v0[0] * inv, v0[1] * inv, v0[2] * inv, v0[3] * inv};
    float4 s1 = {v0[4] * inv, v0[5] * inv, v0[6] * inv, v0[7] * inv};
    float4 s2 = {v1[0] * inv, v1[1] * inv, v1[2] * inv, v1[3] * inv};
    float4 s3 = {v1[4] * inv, v1[5] * inv, v1[6] * inv, v1[7] * inv};
    *(float4*)(xout + row * F + 8 * g)          = s0;
    *(float4*)(xout + row * F + 8 * g + 4)      = s1;
    *(float4*)(xout + row * F + 16 + 8 * g)     = s2;
    *(float4*)(xout + row * F + 16 + 8 * g + 4) = s3;
  }
}

// ---------------- host ----------------
extern "C" void kernel_launch(void* const* d_in, const int* in_sizes, int n_in,
                              void* d_out, int out_size, void* d_ws, size_t ws_size,
                              hipStream_t stream) {
  const float* x      = (const float*)d_in[0];
  const long long* ei = (const long long*)d_in[1];
  const float* ea     = (const float*)d_in[2];
  const float* deg    = (const float*)d_in[3];
  PP pp;
  for (int i = 0; i < 64; ++i) pp.p[i] = (const float*)d_in[4 + i];

  char* w = (char*)d_ws;
  _Float16* wsW = (_Float16*)w;                               // 245760 B
  float* wsB    = (float*)(w + 245760);                       // 7168 B
  float* xbuf0  = (float*)(w + 252928);                       // 3.2 MB
  float* xbuf1  = xbuf0 + (long)NN * F;
  float* agg    = xbuf1 + (long)NN * F;
  float* ebuf0  = agg + (long)NN * F;
  float* ebuf1  = ebuf0 + (long)NE * F;

  conv_weights_kernel<<<40, 256, 0, stream>>>(pp, wsW, wsB);
  pad_feat_kernel<<<((long)NN * F + 255) / 256, 256, 0, stream>>>(x, xbuf0, NN, 16);
  pad_feat_kernel<<<((long)NE * F + 255) / 256, 256, 0, stream>>>(ea, ebuf0, NE, 8);

  const long long* srcI = ei;
  const long long* dstI = ei + NE;
  float* xout_final = (float*)d_out;
  float* eout_final = (float*)d_out + (long)NN * F;

  float* xc = xbuf0;
  float* ec = ebuf0;
  for (int l = 0; l < 4; ++l) {
    float* en = (l == 3) ? eout_final : ((l & 1) ? ebuf0 : ebuf1);
    float* xn = (l == 3) ? xout_final : ((l & 1) ? xbuf0 : xbuf1);
    zero_kernel<<<((long)NN * F + 255) / 256, 256, 0, stream>>>(agg, (long)NN * F);
    edge_mlp_kernel<<<(NE + ROWS - 1) / ROWS, 256, 0, stream>>>(
        xc, ec, srcI, dstI,
        wsW + (size_t)l * LAYER_W_STRIDE, wsB + (size_t)l * LAYER_B_STRIDE,
        en, agg);
    node_mlp_kernel<<<(NN + ROWS - 1) / ROWS, 256, 0, stream>>>(
        xc, agg, deg,
        wsW + (size_t)l * LAYER_W_STRIDE + NW_BASE,
        wsB + (size_t)l * LAYER_B_STRIDE + 224,
        xn);
    xc = xn;
    ec = en;
  }
}
